// RNNDecoder_29953101922991
// MI455X (gfx1250) — matrix operand order repaired and checked by reading, not compiled, not run on hardware
//
#include <hip/hip_runtime.h>
#include <hip/hip_bf16.h>
#include <math.h>

typedef _Float16 f16;
typedef __attribute__((ext_vector_type(16))) _Float16 v16h;
typedef __attribute__((ext_vector_type(8)))  float    v8f;

// Problem constants
#define NB   128     // batch
#define HID  512
#define EMB  80
#define CTX  64
#define LEN  512     // H*W
#define TT   150     // steps
#define TOK  556
#define TOKP 560     // padded to 16
#define SOS  553     // TOKEN_SIZE - 3
#define KX   592     // EMB + HID
#define KXP  608     // padded to 32
#define G3   1536    // 3*HID
#define KHC  576     // HID + CTX

__device__ __forceinline__ float sigmoidf(float x) { return 1.0f / (1.0f + __expf(-x)); }

// ---------------------------------------------------------------------------
// Weight pre-pack: W [Nsrc x Ksrc] f32 row-major  ->  B-fragment-swizzled f16.
// Output layout: [(nt*nkt + kt)*32 + lane]*16 + e, where for the WMMA
// 32x16 B tile: K = kt*32 + (lane>>4)*16 + e, N = nt*16 + (lane&15).
// So the GEMM reads one contiguous 32B chunk per lane per k-step.
// ---------------------------------------------------------------------------
__global__ void pack_b_kernel(const float* __restrict__ W, f16* __restrict__ dst,
                              int Nsrc, int Ksrc, int nkt, int ntiles) {
  long idx = (long)blockIdx.x * blockDim.x + threadIdx.x;
  long total = (long)nkt * ntiles * 512;
  if (idx >= total) return;
  int e    = (int)(idx & 15);
  int lane = (int)((idx >> 4) & 31);
  long tile = idx >> 9;
  int kt = (int)(tile % nkt);
  int nt = (int)(tile / nkt);
  int k  = kt * 32 + (lane >> 4) * 16 + e;
  int nn = nt * 16 + (lane & 15);
  float v = (k < Ksrc && nn < Nsrc) ? W[(long)nn * Ksrc + k] : 0.0f;
  dst[idx] = (f16)v;
}

// encT[n][l][c] = rnn_enc[n][c][l]
__global__ void enc_transpose_kernel(const float* __restrict__ enc, float* __restrict__ encT) {
  long idx = (long)blockIdx.x * blockDim.x + threadIdx.x;
  if (idx >= (long)NB * LEN * CTX) return;
  int c = (int)(idx & 63);
  long rest = idx >> 6;
  int l = (int)(rest & (LEN - 1));
  int n = (int)(rest >> 9);
  encT[idx] = enc[((long)n * CTX + c) * LEN + l];
}

__global__ void zero_kernel(float* __restrict__ p, int count) {
  int i = blockIdx.x * blockDim.x + threadIdx.x;
  if (i < count) p[i] = 0.0f;
}

// ---------------------------------------------------------------------------
// Generic WMMA GEMM: C[M x Nreal] = A[M x nkt*32] (f32) * B (pre-swizzled f16)
// One wave per 16x16 C tile; 4 waves/block on consecutive n-tiles.
// EPI: 0 = none, 1 = +bias[n], 2 = tanh
// ---------------------------------------------------------------------------
template <int EPI, typename OutT>
__global__ void gemm_wmma_kernel(const float* __restrict__ A, long bsA, int lda,
                                 const f16* __restrict__ Bswz, int nkt,
                                 OutT* __restrict__ C, long bsC, int ldc,
                                 int Nreal, int ntiles,
                                 const float* __restrict__ bias) {
  const int wave = threadIdx.x >> 5;
  const int lane = threadIdx.x & 31;
  const int mt   = blockIdx.x;
  const int nt   = blockIdx.y * 4 + wave;
  if (nt >= ntiles) return;
  const int b = blockIdx.z;
  A += (long)b * bsA;
  C += (long)b * bsC;

  const int m0 = mt * 16, n0 = nt * 16;
  // A-fragment addressing (contiguous k = 16*(lane>>4) + e, matching packed B)
  const int am  = lane & 15;
  const int akb = (lane >> 4) * 16;
  const float* arow = A + (long)(m0 + am) * lda;
  // B pre-swizzled: contiguous 16 halfs per lane per k-tile
  const f16* btile = Bswz + (((long)nt * nkt) << 9) + ((long)lane << 4);

  v8f acc = {};
  for (int kt = 0; kt < nkt; ++kt) {
    v16h bb = *(const v16h*)(btile + ((long)kt << 9));
    v16h a;
    const int k0 = kt * 32;
#pragma unroll
    for (int i = 0; i < 8; ++i) {
      int kk = k0 + akb + 2 * i;
      a[2 * i]     = (f16)arow[kk];
      a[2 * i + 1] = (f16)arow[kk + 1];
    }
    acc = __builtin_amdgcn_wmma_f32_16x16x32_f16(false, a, false, bb,
                                                 (short)0, acc, false, false);
  }

  // C/D layout: VGPR v -> M = v + 8*(lane>>4), N = lane&15
  const int cm = (lane >> 4) * 8;
  const int cn = lane & 15;
  if (n0 + cn >= Nreal) return;
  float bv = (EPI == 1) ? bias[n0 + cn] : 0.0f;
#pragma unroll
  for (int v = 0; v < 8; ++v) {
    float val = acc[v];
    if (EPI == 1) val += bv;
    if (EPI == 2) val = tanhf(val);
    C[(long)(m0 + cm + v) * ldc + n0 + cn] = (OutT)val;
  }
}

// x[n][0:80]=embedding[token], x[n][80:592]=out_prev, x[n][592:608]=0
__global__ void build_x_kernel(const float* __restrict__ emb, const int* __restrict__ decoded,
                               const float* __restrict__ outp, float* __restrict__ x, int t) {
  int idx = blockIdx.x * blockDim.x + threadIdx.x;
  if (idx >= NB * KXP) return;
  int n = idx / KXP, j = idx - n * KXP;
  float val;
  if (j < EMB) {
    int tok = (t == 0) ? SOS : decoded[n * TT + (t - 1)];
    val = emb[tok * EMB + j];
  } else if (j < KX) {
    val = outp[n * HID + (j - EMB)];
  } else {
    val = 0.0f;
  }
  x[idx] = val;
}

// GRU gates (r,z,n order); writes new hidden and hc[:, 0:512]
__global__ void gates_kernel(const float* __restrict__ gi, const float* __restrict__ gh,
                             float* __restrict__ hidden, float* __restrict__ hc) {
  int idx = blockIdx.x * blockDim.x + threadIdx.x;
  if (idx >= NB * HID) return;
  int n = idx >> 9, j = idx & (HID - 1);
  const float* gin = gi + (long)n * G3;
  const float* ghn = gh + (long)n * G3;
  float r  = sigmoidf(gin[j] + ghn[j]);
  float z  = sigmoidf(gin[HID + j] + ghn[HID + j]);
  float nn = tanhf(gin[2 * HID + j] + r * ghn[2 * HID + j]);
  float h  = (1.0f - z) * nn + z * hidden[idx];
  hidden[idx] = h;
  hc[(long)n * KHC + j] = h;
}

// score[n][l] = sum_s v[s]*tanh(hW1[n][s] + enc_proj[n][l][s]); 1 wave per (n,l)
__global__ void score_kernel(const float* __restrict__ hW1, const f16* __restrict__ encproj,
                             const float* __restrict__ vvec, float* __restrict__ score) {
  int wid  = (int)(((long)blockIdx.x * blockDim.x + threadIdx.x) >> 5);
  int lane = threadIdx.x & 31;
  if (wid >= NB * LEN) return;
  int n  = wid >> 9;
  int s0 = lane * 16;
  const float* h  = hW1 + (long)n * HID + s0;
  const float* vv = vvec + s0;
  const f16*   ep = encproj + ((long)wid << 9) + s0;
  float acc = 0.0f;
#pragma unroll
  for (int i = 0; i < 16; ++i) acc += vv[i] * tanhf(h[i] + (float)ep[i]);
#pragma unroll
  for (int off = 16; off; off >>= 1) acc += __shfl_xor(acc, off, 32);
  if (lane == 0) score[wid] = acc;
}

// softmax over L=512; one block (256 threads) per batch row
__global__ void softmax_att_kernel(const float* __restrict__ score, float* __restrict__ att) {
  __shared__ float red[256];
  int n = blockIdx.x, tid = threadIdx.x;
  const float* s = score + (long)n * LEN;
  float a = s[tid], b = s[tid + 256];
  red[tid] = fmaxf(a, b);
  __syncthreads();
  for (int o = 128; o; o >>= 1) { if (tid < o) red[tid] = fmaxf(red[tid], red[tid + o]); __syncthreads(); }
  float m = red[0];
  __syncthreads();
  float ea = __expf(a - m), eb = __expf(b - m);
  red[tid] = ea + eb;
  __syncthreads();
  for (int o = 128; o; o >>= 1) { if (tid < o) red[tid] += red[tid + o]; __syncthreads(); }
  float inv = 1.0f / red[0];
  att[(long)n * LEN + tid]       = ea * inv;
  att[(long)n * LEN + tid + 256] = eb * inv;
}

// context[n][c] = sum_l att[n][l]*rnn_enc[n][c][l] -> hc[:, 512:576]; 1 wave per (n,c)
__global__ void context_kernel(const float* __restrict__ att, const float* __restrict__ enc,
                               float* __restrict__ hc) {
  int wid  = (int)(((long)blockIdx.x * blockDim.x + threadIdx.x) >> 5);
  int lane = threadIdx.x & 31;
  if (wid >= NB * CTX) return;
  int n = wid >> 6, c = wid & 63;
  const float* a = att + (long)n * LEN + lane * 16;
  const float* e = enc + ((long)n * CTX + c) * LEN + lane * 16;
  float acc = 0.0f;
#pragma unroll
  for (int i = 0; i < 16; ++i) acc += a[i] * e[i];
#pragma unroll
  for (int off = 16; off; off >>= 1) acc += __shfl_xor(acc, off, 32);
  if (lane == 0) hc[(long)n * KHC + HID + c] = acc;
}

// softmax over 556 logits, scatter into out[n][tok][t] (stride TT)
__global__ void softmax_p_kernel(const float* __restrict__ logits, float* __restrict__ out, int t) {
  __shared__ float red[256];
  int n = blockIdx.x, tid = threadIdx.x;
  const float* lg = logits + (long)n * TOKP;
  float a = lg[tid], b = lg[tid + 256];
  bool hasc = (tid + 512) < TOK;
  float c = hasc ? lg[tid + 512] : -1e30f;
  red[tid] = fmaxf(fmaxf(a, b), c);
  __syncthreads();
  for (int o = 128; o; o >>= 1) { if (tid < o) red[tid] = fmaxf(red[tid], red[tid + o]); __syncthreads(); }
  float m = red[0];
  __syncthreads();
  float ea = __expf(a - m), eb = __expf(b - m), ec = hasc ? __expf(c - m) : 0.0f;
  red[tid] = ea + eb + ec;
  __syncthreads();
  for (int o = 128; o; o >>= 1) { if (tid < o) red[tid] += red[tid + o]; __syncthreads(); }
  float inv = 1.0f / red[0];
  long base = (long)n * TOK * TT + t;
  out[base + (long)tid * TT]         = ea * inv;
  out[base + (long)(tid + 256) * TT] = eb * inv;
  if (hasc) out[base + (long)(tid + 512) * TT] = ec * inv;
}

// ---------------------------------------------------------------------------
extern "C" void kernel_launch(void* const* d_in, const int* in_sizes, int n_in,
                              void* d_out, int out_size, void* d_ws, size_t ws_size,
                              hipStream_t stream) {
  const float* rnn_enc = (const float*)d_in[0];
  const int*   decoded = (const int*)d_in[1];
  const float* emb     = (const float*)d_in[2];
  const float* w_ih    = (const float*)d_in[3];
  const float* w_hh    = (const float*)d_in[4];
  const float* b_ih    = (const float*)d_in[5];
  const float* b_hh    = (const float*)d_in[6];
  const float* W1      = (const float*)d_in[7];
  const float* W2      = (const float*)d_in[8];
  const float* vvec    = (const float*)d_in[9];
  const float* Wc      = (const float*)d_in[10];
  const float* Wo      = (const float*)d_in[11];
  float* out = (float*)d_out;

  // k-tile / n-tile counts
  const int nkt_ih = KXP / 32;   // 19
  const int nkt_hh = HID / 32;   // 16
  const int nkt_w1 = HID / 32;   // 16
  const int nkt_w2 = CTX / 32;   // 2
  const int nkt_wc = KHC / 32;   // 18
  const int nkt_wo = HID / 32;   // 16
  const int nt_g3  = G3 / 16;    // 96
  const int nt_hid = HID / 16;   // 32
  const int nt_tok = TOKP / 16;  // 35

  // Workspace carve-up (256B aligned)
  char* ws = (char*)d_ws;
  size_t off = 0;
  auto alloc = [&](size_t bytes) -> void* {
    void* p = ws + off;
    off = (off + bytes + 255) & ~(size_t)255;
    return p;
  };
  f16* w_ihS  = (f16*)alloc((size_t)nkt_ih * nt_g3  * 512 * 2);
  f16* w_hhS  = (f16*)alloc((size_t)nkt_hh * nt_g3  * 512 * 2);
  f16* W1S    = (f16*)alloc((size_t)nkt_w1 * nt_hid * 512 * 2);
  f16* W2S    = (f16*)alloc((size_t)nkt_w2 * nt_hid * 512 * 2);
  f16* WcS    = (f16*)alloc((size_t)nkt_wc * nt_hid * 512 * 2);
  f16* WoS    = (f16*)alloc((size_t)nkt_wo * nt_tok * 512 * 2);
  f16* encprj = (f16*)alloc((size_t)NB * LEN * HID * 2);       // 67 MB (L2 resident)
  float* encT   = (float*)alloc((size_t)NB * LEN * CTX * 4);
  float* hidden = (float*)alloc((size_t)NB * HID * 4);
  float* outp   = (float*)alloc((size_t)NB * HID * 4);
  float* x      = (float*)alloc((size_t)NB * KXP * 4);
  float* gi     = (float*)alloc((size_t)NB * G3 * 4);
  float* gh     = (float*)alloc((size_t)NB * G3 * 4);
  float* hW1    = (float*)alloc((size_t)NB * HID * 4);
  float* score  = (float*)alloc((size_t)NB * LEN * 4);
  float* att    = (float*)alloc((size_t)NB * LEN * 4);
  float* hc     = (float*)alloc((size_t)NB * KHC * 4);
  float* logits = (float*)alloc((size_t)NB * TOKP * 4);

  auto packB = [&](const float* W, f16* dst, int Nsrc, int Ksrc, int nkt, int ntiles) {
    long total = (long)nkt * ntiles * 512;
    pack_b_kernel<<<(unsigned)((total + 255) / 256), 256, 0, stream>>>(W, dst, Nsrc, Ksrc, nkt, ntiles);
  };

  // ---- one-time prep ----
  packB(w_ih, w_ihS, G3, KX, nkt_ih, nt_g3);
  packB(w_hh, w_hhS, G3, HID, nkt_hh, nt_g3);
  packB(W1,   W1S,   HID, HID, nkt_w1, nt_hid);
  packB(W2,   W2S,   HID, CTX, nkt_w2, nt_hid);
  packB(Wc,   WcS,   HID, KHC, nkt_wc, nt_hid);
  packB(Wo,   WoS,   TOK, HID, nkt_wo, nt_tok);

  {
    long total = (long)NB * LEN * CTX;
    enc_transpose_kernel<<<(unsigned)((total + 255) / 256), 256, 0, stream>>>(rnn_enc, encT);
  }
  // enc_proj[n] = encT[n] (512x64) * W2^T (64x512), stored f16
  gemm_wmma_kernel<0, f16><<<dim3(LEN / 16, nt_hid / 4, NB), 128, 0, stream>>>(
      encT, (long)LEN * CTX, CTX, W2S, nkt_w2,
      encprj, (long)LEN * HID, HID, HID, nt_hid, nullptr);

  zero_kernel<<<(NB * HID + 255) / 256, 256, 0, stream>>>(hidden, NB * HID);
  zero_kernel<<<(NB * HID + 255) / 256, 256, 0, stream>>>(outp, NB * HID);

  // ---- decode loop ----
  const dim3 gemm_blk(128);
  for (int t = 0; t < TT; ++t) {
    build_x_kernel<<<(NB * KXP + 255) / 256, 256, 0, stream>>>(emb, decoded, outp, x, t);

    // gi = x @ w_ih^T + b_ih   (128 x 1536, K=608)
    gemm_wmma_kernel<1, float><<<dim3(NB / 16, nt_g3 / 4, 1), gemm_blk, 0, stream>>>(
        x, 0, KXP, w_ihS, nkt_ih, gi, 0, G3, G3, nt_g3, b_ih);
    // gh = hidden @ w_hh^T + b_hh
    gemm_wmma_kernel<1, float><<<dim3(NB / 16, nt_g3 / 4, 1), gemm_blk, 0, stream>>>(
        hidden, 0, HID, w_hhS, nkt_hh, gh, 0, G3, G3, nt_g3, b_hh);

    gates_kernel<<<(NB * HID + 255) / 256, 256, 0, stream>>>(gi, gh, hidden, hc);

    // hW1 = hidden @ W1^T
    gemm_wmma_kernel<0, float><<<dim3(NB / 16, nt_hid / 4, 1), gemm_blk, 0, stream>>>(
        hidden, 0, HID, W1S, nkt_w1, hW1, 0, HID, HID, nt_hid, nullptr);

    // score / softmax / context
    {
      long waves = (long)NB * LEN;
      score_kernel<<<(unsigned)((waves * 32 + 255) / 256), 256, 0, stream>>>(hW1, encprj, vvec, score);
    }
    softmax_att_kernel<<<NB, 256, 0, stream>>>(score, att);
    {
      long waves = (long)NB * CTX;
      context_kernel<<<(unsigned)((waves * 32 + 255) / 256), 256, 0, stream>>>(att, rnn_enc, hc);
    }

    // out = tanh(hc @ Wc^T)
    gemm_wmma_kernel<2, float><<<dim3(NB / 16, nt_hid / 4, 1), gemm_blk, 0, stream>>>(
        hc, 0, KHC, WcS, nkt_wc, outp, 0, HID, HID, nt_hid, nullptr);

    // logits = out @ Wo^T  (Nreal = 556, padded tiles = 35)
    gemm_wmma_kernel<0, float><<<dim3(NB / 16, (nt_tok + 3) / 4, 1), gemm_blk, 0, stream>>>(
        outp, 0, HID, WoS, nkt_wo, logits, 0, TOKP, TOK, nt_tok, nullptr);

    softmax_p_kernel<<<NB, 256, 0, stream>>>(logits, out, t);
  }
}
